// DMSMHA_Block_28166395527482
// MI455X (gfx1250) — matrix operand order repaired and checked by reading, not compiled, not run on hardware
//
#include <hip/hip_runtime.h>

typedef _Float16 v4h  __attribute__((ext_vector_type(4)));
typedef _Float16 v8h  __attribute__((ext_vector_type(8)));
typedef _Float16 v16h __attribute__((ext_vector_type(16)));
typedef float    v8f  __attribute__((ext_vector_type(8)));

#define ROWS   43520          // B*N == B*P
#define NQ     21760
#define PTOT   21760
#define CDIM   256
#define TM     128
#define TN     64
#define TK     32
#define LDA_S  40              // padded LDS stride in halves (80 B, 16B-aligned)
#define LDB_S  40

// ---------------------------------------------------------------------------
// WMMA GEMM: out[M,N] = epi(A[M,K] @ W[K,N] + bias, resid)
// EPI: 0 = +bias | 1 = relu(+bias) | 2 = +bias+resid
// N, K compile-time so store addressing folds to immediate offsets.
// Double-buffered LDS pipeline: global->reg loads for step i+1 overlap the
// WMMAs of step i; one barrier per K-step.
// ---------------------------------------------------------------------------
template <int EPI, int N, int K>
__global__ __launch_bounds__(256)
void wmma_gemm(const float* __restrict__ A, const float* __restrict__ W,
               const float* __restrict__ bias, const float* __restrict__ resid,
               float* __restrict__ out)
{
    __shared__ __align__(16) _Float16 As[2][TM * LDA_S];
    __shared__ __align__(16) _Float16 Bs[2][TN * LDB_S];   // transposed: Bs[n][k]

    const int tid  = threadIdx.x;
    const int lane = tid & 31;
    const int wid  = tid >> 5;
    const int wm   = wid >> 1;       // 0..3 (32-row strips)
    const int wn   = wid & 1;        // 0..1 (32-col strips)

    const int rowBase = blockIdx.y * TM;
    const int colBase = blockIdx.x * TN;

    // per-thread staging coordinates (compile-time strides)
    // A tile: 128x32 = 4096 elems, 4 float4 per thread
    const int ar[4] = { (0 * 256 + tid) >> 3, (1 * 256 + tid) >> 3,
                        (2 * 256 + tid) >> 3, (3 * 256 + tid) >> 3 };
    const int ac    = (tid & 7) * 4;
    // W tile: 32x64 = 2048 elems, 2 float4 per thread
    const int wk[2] = { (0 * 256 + tid) >> 4, (1 * 256 + tid) >> 4 };
    const int wc    = (tid & 15) * 4;

    float4 aReg[4];
    float4 wReg[2];

    auto loadTiles = [&](int kk) {
        #pragma unroll
        for (int i = 0; i < 4; ++i)
            aReg[i] = *(const float4*)&A[(size_t)(rowBase + ar[i]) * K + kk + ac];
        #pragma unroll
        for (int i = 0; i < 2; ++i)
            wReg[i] = *(const float4*)&W[(size_t)(kk + wk[i]) * N + colBase + wc];
    };
    auto storeTiles = [&](int buf) {
        #pragma unroll
        for (int i = 0; i < 4; ++i) {
            v4h hv = { (_Float16)aReg[i].x, (_Float16)aReg[i].y,
                       (_Float16)aReg[i].z, (_Float16)aReg[i].w };
            *(v4h*)&As[buf][ar[i] * LDA_S + ac] = hv;
        }
        #pragma unroll
        for (int i = 0; i < 2; ++i) {
            Bs[buf][(wc + 0) * LDB_S + wk[i]] = (_Float16)wReg[i].x;
            Bs[buf][(wc + 1) * LDB_S + wk[i]] = (_Float16)wReg[i].y;
            Bs[buf][(wc + 2) * LDB_S + wk[i]] = (_Float16)wReg[i].z;
            Bs[buf][(wc + 3) * LDB_S + wk[i]] = (_Float16)wReg[i].w;
        }
    };

    v8f c[2][2] = {};
    const int rsel = lane & 15;
    const int hi   = lane >> 4;

    constexpr int STEPS = K / TK;

    loadTiles(0);
    storeTiles(0);
    __syncthreads();

    for (int step = 0; step < STEPS; ++step) {
        const int cur = step & 1;
        if (step + 1 < STEPS) loadTiles((step + 1) * TK);
        if (step + 2 < STEPS)   // keep the A stream hot in GL2 (global_prefetch_b8)
            __builtin_prefetch(&A[(size_t)(rowBase + (tid >> 1)) * K +
                                  (step + 2) * TK + (tid & 1) * 16], 0, 1);

        // A fragments (ISA 16-bit A 16x32 layout)
        v16h a[2];
        #pragma unroll
        for (int fm = 0; fm < 2; ++fm) {
            const _Float16* ap = &As[cur][(wm * 32 + fm * 16 + rsel) * LDA_S];
            v8h lo = *(const v8h*)(ap + hi * 8);
            v8h hh = *(const v8h*)(ap + 16 + hi * 8);
            a[fm] = __builtin_shufflevector(lo, hh, 0,1,2,3,4,5,6,7,8,9,10,11,12,13,14,15);
        }
        // B fragments (column = lane&15, K halves split by half-wave)
        v16h b[2];
        #pragma unroll
        for (int fn = 0; fn < 2; ++fn) {
            const _Float16* bp = &Bs[cur][(wn * 32 + fn * 16 + rsel) * LDB_S];
            v8h lo = *(const v8h*)(bp + hi * 8);
            v8h hh = *(const v8h*)(bp + 16 + hi * 8);
            b[fn] = __builtin_shufflevector(lo, hh, 0,1,2,3,4,5,6,7,8,9,10,11,12,13,14,15);
        }
        #pragma unroll
        for (int fm = 0; fm < 2; ++fm)
            #pragma unroll
            for (int fn = 0; fn < 2; ++fn)
                c[fm][fn] = __builtin_amdgcn_wmma_f32_16x16x32_f16(
                    false, a[fm], false, b[fn], (short)0, c[fm][fn], false, false);

        if (step + 1 < STEPS) storeTiles(cur ^ 1);
        __syncthreads();
    }

    // epilogue (ISA f32 C/D layout: vgpr r -> M=r (+8 for hi half-wave), N=lane&15)
    const int colSel = lane & 15;
    const int rOff   = hi * 8;
    #pragma unroll
    for (int fm = 0; fm < 2; ++fm) {
        #pragma unroll
        for (int fn = 0; fn < 2; ++fn) {
            const int col = colBase + wn * 32 + fn * 16 + colSel;
            const float bv = bias[col];
            const int row0 = rowBase + wm * 32 + fm * 16 + rOff;
            float* po = out + (size_t)row0 * N + col;
            const float* pr = resid + (size_t)row0 * N + col;
            #pragma unroll
            for (int r = 0; r < 8; ++r) {
                float val = c[fm][fn][r] + bv;
                if (EPI == 1) val = val > 0.f ? val : 0.f;
                if (EPI == 2) val += pr[r * N];
                po[r * N] = val;
            }
        }
    }
}

// ---------------------------------------------------------------------------
// softmax over 16 logits per (b,n,h), in place
// ---------------------------------------------------------------------------
__global__ __launch_bounds__(256)
void softmax16(float* __restrict__ aw, int total)
{
    int t = blockIdx.x * 256 + threadIdx.x;
    if (t >= total) return;
    float* p = aw + (size_t)t * 16;
    float m = -1e30f;
    #pragma unroll
    for (int i = 0; i < 16; ++i) m = fmaxf(m, p[i]);
    float e[16], s = 0.f;
    #pragma unroll
    for (int i = 0; i < 16; ++i) { e[i] = __expf(p[i] - m); s += e[i]; }
    const float inv = 1.f / s;
    #pragma unroll
    for (int i = 0; i < 16; ++i) p[i] = e[i] * inv;
}

// ---------------------------------------------------------------------------
// MSDA bilinear gather: block=(32,8): lane = channel d, y = head h.
// value:(B,P,H,32)  off:(rows, H*L*K*2)  aw:(rows, H*16)  ref:(rows, L, 2)
// out:(rows, 256) with channel = h*32 + d
// ---------------------------------------------------------------------------
__global__ __launch_bounds__(256)
void msda_sample(const float* __restrict__ value, const float* __restrict__ off,
                 const float* __restrict__ aw, const float* __restrict__ ref,
                 float* __restrict__ out)
{
    const int d   = threadIdx.x;     // 0..31
    const int h   = threadIdx.y;     // 0..7
    const int row = blockIdx.x;      // b*NQ + n
    const int b   = row / NQ;

    constexpr int HL[4] = {128, 64, 32, 16};
    constexpr int WL[4] = {128, 64, 32, 16};
    constexpr int S0[4] = {0, 16384, 20480, 21504};

    const float* op = off + ((size_t)row * 8 + h) * 32;   // L*K*2 per head
    const float* ap = aw  + ((size_t)row * 8 + h) * 16;

    float acc = 0.f;
    #pragma unroll
    for (int l = 0; l < 4; ++l) {
        const float rx = ref[((size_t)row * 4 + l) * 2 + 0];
        const float ry = ref[((size_t)row * 4 + l) * 2 + 1];
        const int Hl = HL[l], Wl = WL[l], s0 = S0[l];
        #pragma unroll
        for (int k = 0; k < 4; ++k) {
            const float x = rx * (float)Wl + op[(l * 4 + k) * 2 + 0] - 0.5f;
            const float y = ry * (float)Hl + op[(l * 4 + k) * 2 + 1] - 0.5f;
            const float xf = floorf(x), yf = floorf(y);
            const float wx = x - xf, wy = y - yf;
            const int x0 = (int)xf, y0 = (int)yf;
            float s = 0.f;
            #pragma unroll
            for (int cy = 0; cy < 2; ++cy) {
                #pragma unroll
                for (int cx = 0; cx < 2; ++cx) {
                    const int xi = x0 + cx, yi = y0 + cy;
                    if (xi >= 0 && xi < Wl && yi >= 0 && yi < Hl) {
                        const float cw = (cx ? wx : 1.f - wx) * (cy ? wy : 1.f - wy);
                        const size_t vi =
                            (((size_t)b * PTOT + s0 + yi * Wl + xi) * 8 + h) * 32 + d;
                        s += cw * value[vi];
                    }
                }
            }
            acc += ap[l * 4 + k] * s;
        }
    }
    out[(size_t)row * CDIM + h * 32 + d] = acc;
}

// ---------------------------------------------------------------------------
// out = LayerNorm(y + rscale*resid) * g + be   (block per row, C=256)
// ---------------------------------------------------------------------------
__global__ __launch_bounds__(256)
void add_ln(const float* __restrict__ y, const float* __restrict__ resid, float rscale,
            const float* __restrict__ g, const float* __restrict__ be,
            float* __restrict__ out)
{
    __shared__ float red[256];
    const int row = blockIdx.x, c = threadIdx.x;
    const size_t o = (size_t)row * CDIM + c;
    const float t = y[o] + rscale * resid[o];

    red[c] = t; __syncthreads();
    #pragma unroll
    for (int s = 128; s > 0; s >>= 1) { if (c < s) red[c] += red[c + s]; __syncthreads(); }
    const float mean = red[0] * (1.f / 256.f);
    __syncthreads();

    const float dx = t - mean;
    red[c] = dx * dx; __syncthreads();
    #pragma unroll
    for (int s = 128; s > 0; s >>= 1) { if (c < s) red[c] += red[c + s]; __syncthreads(); }
    const float var = red[0] * (1.f / 256.f);

    out[o] = dx * rsqrtf(var + 1e-5f) * g[c] + be[c];
}

// ---------------------------------------------------------------------------
extern "C" void kernel_launch(void* const* d_in, const int* in_sizes, int n_in,
                              void* d_out, int out_size, void* d_ws, size_t ws_size,
                              hipStream_t stream)
{
    (void)in_sizes; (void)n_in; (void)out_size; (void)ws_size;
    const float* q      = (const float*)d_in[0];
    const float* v      = (const float*)d_in[2];
    const float* refpts = (const float*)d_in[3];
    const float* W_off  = (const float*)d_in[6];
    const float* b_off  = (const float*)d_in[7];
    const float* W_attn = (const float*)d_in[8];
    const float* b_attn = (const float*)d_in[9];
    const float* W_val  = (const float*)d_in[10];
    const float* b_val  = (const float*)d_in[11];
    const float* W_out  = (const float*)d_in[12];
    const float* b_out  = (const float*)d_in[13];
    const float* W1     = (const float*)d_in[14];
    const float* b1     = (const float*)d_in[15];
    const float* W2     = (const float*)d_in[16];
    const float* b2     = (const float*)d_in[17];
    const float* g1     = (const float*)d_in[18];
    const float* be1    = (const float*)d_in[19];
    const float* g2     = (const float*)d_in[20];
    const float* be2    = (const float*)d_in[21];

    float* ws = (float*)d_ws;
    const size_t S = (size_t)ROWS * CDIM;          // 11,141,120 floats
    float* val  = ws;                  // value projection      (ROWS x 256)
    float* offb = ws + S;              // sampling offsets      (ROWS x 256) -> later x
    float* awb  = ws + 2 * S;          // attn logits/weights   (ROWS x 128)
    float* samp = ws + 2 * S + S / 2;  // sampled output        (ROWS x 256) -> later h
    float* hid  = ws + 3 * S + S / 2;  // ffn hidden            (ROWS x 1024)
    float* attn = val;                 // reuse after sampling
    float* x    = offb;                // reuse after sampling
    float* hbuf = samp;                // reuse after out proj

    const dim3 blk(256);
    const int MB = ROWS / TM;          // 340

    // value = v @ W_val + b_val
    wmma_gemm<0, 256, 256><<<dim3(256 / TN, MB), blk, 0, stream>>>(v, W_val, b_val, nullptr, val);
    // off = q @ W_off + b_off
    wmma_gemm<0, 256, 256><<<dim3(256 / TN, MB), blk, 0, stream>>>(q, W_off, b_off, nullptr, offb);
    // logits = q @ W_attn + b_attn
    wmma_gemm<0, 128, 256><<<dim3(128 / TN, MB), blk, 0, stream>>>(q, W_attn, b_attn, nullptr, awb);
    // softmax over L*K per (b,n,h)
    softmax16<<<dim3((ROWS * 8 + 255) / 256), blk, 0, stream>>>(awb, ROWS * 8);
    // deformable bilinear sampling
    msda_sample<<<dim3(ROWS), dim3(32, 8), 0, stream>>>(val, offb, awb, refpts, samp);
    // attn = samp @ W_out + b_out    (overwrites val)
    wmma_gemm<0, 256, 256><<<dim3(256 / TN, MB), blk, 0, stream>>>(samp, W_out, b_out, nullptr, attn);
    // x = LN(attn + 2*q)             (overwrites offb)
    add_ln<<<dim3(ROWS), blk, 0, stream>>>(attn, q, 2.f, g1, be1, x);
    // hid = relu(x @ W1 + b1)
    wmma_gemm<1, 1024, 256><<<dim3(1024 / TN, MB), blk, 0, stream>>>(x, W1, b1, nullptr, hid);
    // h = hid @ W2 + b2 + x          (overwrites samp)
    wmma_gemm<2, 256, 1024><<<dim3(256 / TN, MB), blk, 0, stream>>>(hid, W2, b2, x, hbuf);
    // out = LN(h)
    add_ln<<<dim3(ROWS), blk, 0, stream>>>(hbuf, hbuf, 0.f, g2, be2, (float*)d_out);
}
